// MultiHeadAttention_37538014167736
// MI455X (gfx1250) — matrix operand order repaired and checked by reading, not compiled, not run on hardware
//
#include <hip/hip_runtime.h>
#include <hip/hip_bf16.h>
#include <math.h>

#define DMODEL 1024
#define NHEADS 16
#define DK     64
#define BATCH  2
#define SEQ    2048
#define MROWS  (BATCH * SEQ)   // 4096

typedef __bf16 bf16_t;
typedef __attribute__((ext_vector_type(16))) __bf16 v16bf;
typedef __attribute__((ext_vector_type(8)))  float  v8f;

union F32x8   { v8f v; float e[8]; };
union BF16x16 { v16bf v; uint4 q[2]; bf16_t e[16]; };

__device__ __forceinline__ v8f wmma_bf16(v16bf a, v16bf b, v8f c) {
  // D = A(16x32 bf16) x B(32x16 bf16) + C(16x16 f32)
  return __builtin_amdgcn_wmma_f32_16x16x32_bf16(
      /*neg_a=*/false, a, /*neg_b=*/false, b,
      /*c_mod=*/(short)0, c, /*reuse_a=*/false, /*reuse_b=*/false);
}

// A operand (16x32 bf16) from row-major memory, row stride ld (elements).
// ISA layout: lane%16 = M row; lanes0-15 hold K {0..7,16..23}, lanes16-31 hold K {8..15,24..31}.
__device__ __forceinline__ v16bf load_a(const bf16_t* base, int ld, int lane) {
  const bf16_t* p = base + (size_t)(lane & 15) * ld + ((lane >> 4) << 3);
  BF16x16 a;
  a.q[0] = *(const uint4*)(p);        // K kb .. kb+7
  a.q[1] = *(const uint4*)(p + 16);   // K kb+16 .. kb+23
  return a.v;
}

// B operand (32x16 bf16) from row-major KxN memory, row stride ld.
// ISA layout: lane%16 = N col; lanes0-15 hold K {0..7,16..23}, lanes16-31 hold K {8..15,24..31}.
__device__ __forceinline__ v16bf load_b(const bf16_t* base, int ld, int lane) {
  const bf16_t* p = base + (size_t)((lane >> 4) << 3) * ld + (lane & 15);
  BF16x16 b;
#pragma unroll
  for (int i = 0; i < 8; ++i) {
    b.e[i]     = p[(size_t)i * ld];
    b.e[8 + i] = p[(size_t)(16 + i) * ld];
  }
  return b.v;
}

// A operand built from f32 row-major memory (converted to bf16 on the fly).
__device__ __forceinline__ v16bf load_a_f32(const float* base, int ld, int lane) {
  const float* p = base + (size_t)(lane & 15) * ld + ((lane >> 4) << 3);
  float4 f0 = *(const float4*)(p);
  float4 f1 = *(const float4*)(p + 4);
  float4 f2 = *(const float4*)(p + 16);
  float4 f3 = *(const float4*)(p + 20);
  BF16x16 a;
  a.e[0]  = (bf16_t)f0.x; a.e[1]  = (bf16_t)f0.y; a.e[2]  = (bf16_t)f0.z; a.e[3]  = (bf16_t)f0.w;
  a.e[4]  = (bf16_t)f1.x; a.e[5]  = (bf16_t)f1.y; a.e[6]  = (bf16_t)f1.z; a.e[7]  = (bf16_t)f1.w;
  a.e[8]  = (bf16_t)f2.x; a.e[9]  = (bf16_t)f2.y; a.e[10] = (bf16_t)f2.z; a.e[11] = (bf16_t)f2.w;
  a.e[12] = (bf16_t)f3.x; a.e[13] = (bf16_t)f3.y; a.e[14] = (bf16_t)f3.z; a.e[15] = (bf16_t)f3.w;
  return a.v;
}

__device__ __forceinline__ float rmax16(float x) {
  x = fmaxf(x, __shfl_xor(x, 1, 32));
  x = fmaxf(x, __shfl_xor(x, 2, 32));
  x = fmaxf(x, __shfl_xor(x, 4, 32));
  x = fmaxf(x, __shfl_xor(x, 8, 32));
  return x;
}
__device__ __forceinline__ float rsum16(float x) {
  x += __shfl_xor(x, 1, 32);
  x += __shfl_xor(x, 2, 32);
  x += __shfl_xor(x, 4, 32);
  x += __shfl_xor(x, 8, 32);
  return x;
}

// ---------------- Kernel A: weight f32 -> bf16 (1M elems, 4/thread) ----------------
__global__ __launch_bounds__(256) void cvt_w_kernel(const float* __restrict__ w,
                                                    bf16_t* __restrict__ o) {
  int i = (blockIdx.x * 256 + threadIdx.x) * 4;
  float4 f = *(const float4*)(w + i);
  union { bf16_t e[4]; uint2 u; } pk;
  pk.e[0] = (bf16_t)f.x; pk.e[1] = (bf16_t)f.y; pk.e[2] = (bf16_t)f.z; pk.e[3] = (bf16_t)f.w;
  *(uint2*)(o + i) = pk.u;
}

// one k-step of operands for a 32x64 GEMM tile
struct KStep { v16bf a0, a1, b0, b1, b2, b3; };

// ------------- Kernel B: fused QKV projections, X[4096,1024] @ W[1024,1024] + b ------------
// Per wave: 32(M) x 64(N) tile; k-loop unrolled x2 with ping-pong tile structs (no reg copies).
// blockIdx.y: 0=q (scaled 1/8, [B,H,S,64]), 1=k (transposed [B,H,64,S]), 2=v ([B,H,S,64])
__global__ __launch_bounds__(256) void qkv_proj_kernel(
    const float* __restrict__ Q, const float* __restrict__ K, const float* __restrict__ V,
    const bf16_t* __restrict__ WqB, const bf16_t* __restrict__ WkB, const bf16_t* __restrict__ WvB,
    const float* __restrict__ bq, const float* __restrict__ bk, const float* __restrict__ bv,
    bf16_t* __restrict__ qbuf, bf16_t* __restrict__ kT, bf16_t* __restrict__ vbuf) {
  const int lane = threadIdx.x & 31;
  const int task = blockIdx.x * 8 + (threadIdx.x >> 5);   // 0..2047
  const int m0 = (task >> 4) * 32;                        // M block (32 rows)
  const int n0 = (task & 15) * 64;                        // N block (64 cols)
  const int which = blockIdx.y;

  const float*  X    = (which == 0) ? Q   : (which == 1) ? K   : V;
  const bf16_t* W    = (which == 0) ? WqB : (which == 1) ? WkB : WvB;
  const float*  bias = (which == 0) ? bq  : (which == 1) ? bk  : bv;

  F32x8 acc[2][4];
#pragma unroll
  for (int mi = 0; mi < 2; ++mi)
#pragma unroll
    for (int t = 0; t < 4; ++t)
#pragma unroll
      for (int i = 0; i < 8; ++i) acc[mi][t].e[i] = 0.0f;

  const float* A0 = X + (size_t)m0 * DMODEL;
  const float* A1 = X + (size_t)(m0 + 16) * DMODEL;

  auto loadStep = [&](KStep& s, int k) {
    s.a0 = load_a_f32(A0 + k, DMODEL, lane);
    s.a1 = load_a_f32(A1 + k, DMODEL, lane);
    const bf16_t* wb = W + (size_t)k * DMODEL + n0;
    s.b0 = load_b(wb,      DMODEL, lane);
    s.b1 = load_b(wb + 16, DMODEL, lane);
    s.b2 = load_b(wb + 32, DMODEL, lane);
    s.b3 = load_b(wb + 48, DMODEL, lane);
  };
  auto compute = [&](KStep& s) {
    acc[0][0].v = wmma_bf16(s.a0, s.b0, acc[0][0].v);
    acc[0][1].v = wmma_bf16(s.a0, s.b1, acc[0][1].v);
    acc[0][2].v = wmma_bf16(s.a0, s.b2, acc[0][2].v);
    acc[0][3].v = wmma_bf16(s.a0, s.b3, acc[0][3].v);
    acc[1][0].v = wmma_bf16(s.a1, s.b0, acc[1][0].v);
    acc[1][1].v = wmma_bf16(s.a1, s.b1, acc[1][1].v);
    acc[1][2].v = wmma_bf16(s.a1, s.b2, acc[1][2].v);
    acc[1][3].v = wmma_bf16(s.a1, s.b3, acc[1][3].v);
  };

  KStep sA, sB;
  loadStep(sA, 0);
  int k = 0;
  for (; k < DMODEL - 64; k += 64) {
    loadStep(sB, k + 32);
    compute(sA);
    loadStep(sA, k + 64);
    compute(sB);
  }
  loadStep(sB, k + 32);   // k == DMODEL-64 here
  compute(sA);
  compute(sB);

  const int hl = (lane >> 4) << 3;
  const int nl = lane & 15;
#pragma unroll
  for (int mi = 0; mi < 2; ++mi) {
#pragma unroll
    for (int t = 0; t < 4; ++t) {
#pragma unroll
      for (int r = 0; r < 8; ++r) {
        int m = m0 + mi * 16 + r + hl;       // global row = b*2048 + s
        int j = n0 + t * 16 + nl;            // model col = h*64 + dk
        float val = acc[mi][t].e[r] + bias[j];
        int bb = m >> 11, s = m & 2047, h = j >> 6, dk = j & 63;
        size_t bh = (size_t)(bb * NHEADS + h);
        if (which == 0)
          qbuf[(bh * SEQ + s) * DK + dk] = (bf16_t)(val * 0.125f);  // pre-scaled by 1/sqrt(64)
        else if (which == 1)
          kT[(bh * DK + dk) * SEQ + s] = (bf16_t)val;                // transposed for B-operand
        else
          vbuf[(bh * SEQ + s) * DK + dk] = (bf16_t)val;
      }
    }
  }
}

// ---------------- Kernel C: flash attention (one wave = one 16-row q block) ----------------
struct KTiles { v16bf k0, k1, k2, k3; };

__global__ __launch_bounds__(256) void attn_kernel(
    const bf16_t* __restrict__ qbuf, const bf16_t* __restrict__ kT,
    const bf16_t* __restrict__ vbuf, bf16_t* __restrict__ Obuf) {
  __shared__ bf16_t ldsP[8 * 16 * 32];   // 8 waves x (16x32 bf16 P tile) = 8KB
  const int lane = threadIdx.x & 31;
  const int wid  = threadIdx.x >> 5;
  const int task = blockIdx.x * 8 + wid;  // 0..4095 ; 8 waves of a block share one head's K/V
  const int bh = task >> 7;               // 0..31  (b*16+h)
  const int s0 = (task & 127) * 16;       // query row block

  const bf16_t* qbase = qbuf + ((size_t)bh * SEQ + s0) * DK;
  const bf16_t* kTb   = kT   + (size_t)bh * DK * SEQ;
  const bf16_t* vb    = vbuf + (size_t)bh * SEQ * DK;
  bf16_t* Pl = ldsP + wid * 512;

  // q tile resident in registers: dk chunks 0-31 and 32-63
  v16bf aQ0 = load_a(qbase,      DK, lane);
  v16bf aQ1 = load_a(qbase + 32, DK, lane);

  float mrow[8], lrow[8];
  F32x8 acc[4];
#pragma unroll
  for (int r = 0; r < 8; ++r) { mrow[r] = -INFINITY; lrow[r] = 0.0f; }
#pragma unroll
  for (int t = 0; t < 4; ++t)
#pragma unroll
    for (int i = 0; i < 8; ++i) acc[t].e[i] = 0.0f;

  const int hl = (lane >> 4) << 3;   // row offset of this lane's C slots
  const int nl = lane & 15;

  auto loadK = [&](KTiles& t, int kb) {
    t.k0 = load_b(kTb + kb,                 SEQ, lane);  // dk 0-31,  keys kb..kb+15
    t.k1 = load_b(kTb + 32 * SEQ + kb,      SEQ, lane);  // dk 32-63, keys kb..kb+15
    t.k2 = load_b(kTb + kb + 16,            SEQ, lane);  // dk 0-31,  keys kb+16..kb+31
    t.k3 = load_b(kTb + 32 * SEQ + kb + 16, SEQ, lane);  // dk 32-63, keys kb+16..kb+31
  };

  auto step = [&](KTiles& t, int kb) {
    // V B-tiles issued first so their latency hides under the softmax math
    v16bf bV0 = load_b(vb + (size_t)kb * DK,      DK, lane);
    v16bf bV1 = load_b(vb + (size_t)kb * DK + 16, DK, lane);
    v16bf bV2 = load_b(vb + (size_t)kb * DK + 32, DK, lane);
    v16bf bV3 = load_b(vb + (size_t)kb * DK + 48, DK, lane);

    // scores: two 16x16 tiles, K=dk=64 in two WMMA steps each
    F32x8 sc0, sc1;
#pragma unroll
    for (int i = 0; i < 8; ++i) { sc0.e[i] = 0.0f; sc1.e[i] = 0.0f; }
    sc0.v = wmma_bf16(aQ0, t.k0, sc0.v);
    sc0.v = wmma_bf16(aQ1, t.k1, sc0.v);
    sc1.v = wmma_bf16(aQ0, t.k2, sc1.v);
    sc1.v = wmma_bf16(aQ1, t.k3, sc1.v);

    // online softmax: slot r of this lane <-> row (r + hl)
#pragma unroll
    for (int r = 0; r < 8; ++r) {
      float rm = rmax16(fmaxf(sc0.e[r], sc1.e[r]));
      float mn = fmaxf(mrow[r], rm);
      float sc = __expf(mrow[r] - mn);
      float p0 = __expf(sc0.e[r] - mn);
      float p1 = __expf(sc1.e[r] - mn);
      lrow[r] = lrow[r] * sc + rsum16(p0 + p1);
      mrow[r] = mn;
      acc[0].e[r] *= sc; acc[1].e[r] *= sc; acc[2].e[r] *= sc; acc[3].e[r] *= sc;
      int row = r + hl;
      Pl[row * 32 + nl]      = (bf16_t)p0;
      Pl[row * 32 + 16 + nl] = (bf16_t)p1;
    }
    asm volatile("s_wait_dscnt 0x0" ::: "memory");   // P tile visible to whole wave

    // P(16x32) @ V(32x64): reload P in A-operand layout
    v16bf aP = load_a(Pl, 32, lane);
    acc[0].v = wmma_bf16(aP, bV0, acc[0].v);
    acc[1].v = wmma_bf16(aP, bV1, acc[1].v);
    acc[2].v = wmma_bf16(aP, bV2, acc[2].v);
    acc[3].v = wmma_bf16(aP, bV3, acc[3].v);
    asm volatile("s_wait_dscnt 0x0" ::: "memory");   // done with P before next overwrite
  };

  KTiles tA, tB;
  loadK(tA, 0);
  int kb = 0;
  for (; kb < SEQ - 64; kb += 64) {
    loadK(tB, kb + 32);
    step(tA, kb);
    loadK(tA, kb + 64);
    step(tB, kb + 32);
  }
  loadK(tB, kb + 32);   // kb == SEQ-64 here
  step(tA, kb);
  step(tB, kb + 32);

  // normalize and write context in [B*S, 1024] row-major (b = bh>>4, h = bh&15)
#pragma unroll
  for (int r = 0; r < 8; ++r) {
    float inv = 1.0f / lrow[r];
    size_t row = (size_t)(bh >> 4) * SEQ + s0 + r + hl;
#pragma unroll
    for (int t = 0; t < 4; ++t)
      Obuf[row * DMODEL + (bh & 15) * DK + t * 16 + nl] = (bf16_t)(acc[t].e[r] * inv);
  }
}

// ---------------- Kernel D: output projection O[4096,1024] @ Wo + bo -> f32 ----------------
// Per wave: 32(M) x 64(N) tile, ping-pong pipelined like the QKV GEMM.
__global__ __launch_bounds__(256) void oproj_kernel(
    const bf16_t* __restrict__ Obuf, const bf16_t* __restrict__ WoB,
    const float* __restrict__ bo, float* __restrict__ out) {
  const int lane = threadIdx.x & 31;
  const int task = blockIdx.x * 8 + (threadIdx.x >> 5);   // 0..2047
  const int m0 = (task >> 4) * 32;
  const int n0 = (task & 15) * 64;

  F32x8 acc[2][4];
#pragma unroll
  for (int mi = 0; mi < 2; ++mi)
#pragma unroll
    for (int t = 0; t < 4; ++t)
#pragma unroll
      for (int i = 0; i < 8; ++i) acc[mi][t].e[i] = 0.0f;

  const bf16_t* A0 = Obuf + (size_t)m0 * DMODEL;
  const bf16_t* A1 = Obuf + (size_t)(m0 + 16) * DMODEL;

  auto loadStep = [&](KStep& s, int k) {
    s.a0 = load_a(A0 + k, DMODEL, lane);
    s.a1 = load_a(A1 + k, DMODEL, lane);
    const bf16_t* wb = WoB + (size_t)k * DMODEL + n0;
    s.b0 = load_b(wb,      DMODEL, lane);
    s.b1 = load_b(wb + 16, DMODEL, lane);
    s.b2 = load_b(wb + 32, DMODEL, lane);
    s.b3 = load_b(wb + 48, DMODEL, lane);
  };
  auto compute = [&](KStep& s) {
    acc[0][0].v = wmma_bf16(s.a0, s.b0, acc[0][0].v);
    acc[0][1].v = wmma_bf16(s.a0, s.b1, acc[0][1].v);
    acc[0][2].v = wmma_bf16(s.a0, s.b2, acc[0][2].v);
    acc[0][3].v = wmma_bf16(s.a0, s.b3, acc[0][3].v);
    acc[1][0].v = wmma_bf16(s.a1, s.b0, acc[1][0].v);
    acc[1][1].v = wmma_bf16(s.a1, s.b1, acc[1][1].v);
    acc[1][2].v = wmma_bf16(s.a1, s.b2, acc[1][2].v);
    acc[1][3].v = wmma_bf16(s.a1, s.b3, acc[1][3].v);
  };

  KStep sA, sB;
  loadStep(sA, 0);
  int k = 0;
  for (; k < DMODEL - 64; k += 64) {
    loadStep(sB, k + 32);
    compute(sA);
    loadStep(sA, k + 64);
    compute(sB);
  }
  loadStep(sB, k + 32);   // k == DMODEL-64 here
  compute(sA);
  compute(sB);

  const int hl = (lane >> 4) << 3;
  const int nl = lane & 15;
#pragma unroll
  for (int mi = 0; mi < 2; ++mi) {
#pragma unroll
    for (int t = 0; t < 4; ++t) {
#pragma unroll
      for (int r = 0; r < 8; ++r) {
        int m = m0 + mi * 16 + r + hl;
        int j = n0 + t * 16 + nl;
        out[(size_t)m * DMODEL + j] = acc[mi][t].e[r] + bo[j];
      }
    }
  }
}

extern "C" void kernel_launch(void* const* d_in, const int* in_sizes, int n_in,
                              void* d_out, int out_size, void* d_ws, size_t ws_size,
                              hipStream_t stream) {
  const float* Q  = (const float*)d_in[0];
  const float* K  = (const float*)d_in[1];
  const float* V  = (const float*)d_in[2];
  const float* Wq = (const float*)d_in[3];
  const float* bq = (const float*)d_in[4];
  const float* Wk = (const float*)d_in[5];
  const float* bk = (const float*)d_in[6];
  const float* Wv = (const float*)d_in[7];
  const float* bv = (const float*)d_in[8];
  const float* Wo = (const float*)d_in[9];
  const float* bo = (const float*)d_in[10];

  char* ws = (char*)d_ws;
  const size_t WBYTES = (size_t)DMODEL * DMODEL * sizeof(bf16_t);        // 2 MB
  const size_t ABYTES = (size_t)MROWS * DMODEL * sizeof(bf16_t);         // 8 MB
  bf16_t* WqB  = (bf16_t*)(ws);
  bf16_t* WkB  = (bf16_t*)(ws + WBYTES);
  bf16_t* WvB  = (bf16_t*)(ws + 2 * WBYTES);
  bf16_t* WoB  = (bf16_t*)(ws + 3 * WBYTES);
  bf16_t* qbuf = (bf16_t*)(ws + 4 * WBYTES);
  bf16_t* kT   = (bf16_t*)(ws + 4 * WBYTES + ABYTES);
  bf16_t* vbuf = (bf16_t*)(ws + 4 * WBYTES + 2 * ABYTES);
  bf16_t* Obuf = (bf16_t*)(ws + 4 * WBYTES + 3 * ABYTES);                // total 40 MB

  // 1) weights -> bf16
  cvt_w_kernel<<<1024, 256, 0, stream>>>(Wq, WqB);
  cvt_w_kernel<<<1024, 256, 0, stream>>>(Wk, WkB);
  cvt_w_kernel<<<1024, 256, 0, stream>>>(Wv, WvB);
  cvt_w_kernel<<<1024, 256, 0, stream>>>(Wo, WoB);

  // 2) QKV projections (256 blocks x 8 waves = 2048 wave-tiles of 32x64 per matrix)
  qkv_proj_kernel<<<dim3(256, 3), 256, 0, stream>>>(Q, K, V, WqB, WkB, WvB,
                                                    bq, bk, bv, qbuf, kT, vbuf);

  // 3) flash attention (4096 wave-tasks: 32 heads x 128 q-blocks)
  attn_kernel<<<512, 256, 0, stream>>>(qbuf, kT, vbuf, Obuf);

  // 4) output projection
  oproj_kernel<<<256, 256, 0, stream>>>(Obuf, WoB, bo, (float*)d_out);
}
